// SkeletonGNN_25391846653993
// MI455X (gfx1250) — compile-verified, matrix operand order checked by reading, not run on hardware
//
#include <hip/hip_runtime.h>

// ---------------- types ----------------
typedef __attribute__((ext_vector_type(16))) __bf16 v16bf;
typedef __attribute__((ext_vector_type(8)))  float  v8f;

union AFrag {
    v16bf v;
    unsigned short u[16];
    unsigned int   u32[8];
};

__device__ __forceinline__ unsigned short f2bf(float f) {
    unsigned int u = __float_as_uint(f);
    u += 0x7FFFu + ((u >> 16) & 1u);          // round-to-nearest-even
    return (unsigned short)(u >> 16);
}

__device__ __forceinline__ void atomicMaxF(float* a, float v) {
    // monotonic bit trick: works with -inf init (0xFF800000)
    if (v >= 0.0f) atomicMax((int*)a, __float_as_int(v));
    else           atomicMin((unsigned int*)a, __float_as_uint(v));
}

// ---------------- utility kernels ----------------
__global__ void k_fill(float* p, float v, long n) {
    long i = (long)blockIdx.x * blockDim.x + threadIdx.x;
    if (i < n) p[i] = v;
}

__global__ void k_f32_to_bf16(const float* __restrict__ in,
                              unsigned short* __restrict__ out, long n) {
    long i = (long)blockIdx.x * blockDim.x + threadIdx.x;
    if (i < n) out[i] = f2bf(in[i]);
}

// convert + transpose: in is [K, NOUT] row-major f32, out is [NOUT, K] bf16.
// Makes B-fragment loads contiguous (2 x b128 per fragment, same as A).
__global__ void k_f32_to_bf16_T(const float* __restrict__ in,
                                unsigned short* __restrict__ out,
                                int K, int NOUT) {
    long t = (long)blockIdx.x * blockDim.x + threadIdx.x;
    if (t >= (long)K * NOUT) return;
    int k = (int)(t / NOUT);
    int n = (int)(t % NOUT);
    out[(long)n * K + k] = f2bf(in[t]);
}

__global__ void k_relu_to_bf16(const float* __restrict__ in,
                               unsigned short* __restrict__ out, long n) {
    long i = (long)blockIdx.x * blockDim.x + threadIdx.x;
    if (i < n) out[i] = f2bf(fmaxf(in[i], 0.0f));
}

// scatter-add of 128-wide features along edges: agg[dst] += feat[src]
__global__ void k_edge_scatter128(const float* __restrict__ feat,
                                  const int* __restrict__ src,
                                  const int* __restrict__ dst,
                                  float* __restrict__ agg, int E) {
    long t = (long)blockIdx.x * blockDim.x + threadIdx.x;
    if (t >= (long)E * 128) return;
    int e = (int)(t >> 7);
    int f = (int)(t & 127);
    atomicAdd(&agg[(long)dst[e] * 128 + f], feat[(long)src[e] * 128 + f]);
}

// ---------------- WMMA GEMM ----------------
// out[nrows, NOUT] = act( A@W (+ A2@W2) + bias (+ res) ), bf16 inputs, f32 acc.
// Weights pre-transposed to [NOUT, K] so A and B fragments load identically
// (2 x dwordx4 per fragment). Each wave computes NT 16x16 tiles sharing one
// A fragment per k-step. Fragment layout per CDNA5 ISA 7.12.2 (bf16).
template<int K, int NOUT, int NT, bool DUAL, bool RELU, bool HASRES, bool STOREBF>
__global__ __launch_bounds__(128)
void k_wmma_gemm(const unsigned short* __restrict__ A,
                 const unsigned short* __restrict__ Wt,
                 const unsigned short* __restrict__ A2,
                 const unsigned short* __restrict__ W2t,
                 const float* __restrict__ bias,
                 const float* __restrict__ res,
                 float* __restrict__ out,
                 unsigned short* __restrict__ outbf,
                 int nrows)
{
    const int lane = threadIdx.x;                 // 0..31 (wave32)
    const int m0   = blockIdx.x * 16;
    const int n0   = (blockIdx.y * 4 + threadIdx.y) * (16 * NT);
    const int mrow = m0 + (lane & 15);
    const int mld  = mrow < nrows ? mrow : nrows - 1;   // keep EXEC uniform
    const int lcol = lane & 15;
    const int kb   = (lane >> 4) * 8;             // lanes 16-31: K offset +8

    v8f acc[NT] = {};

    for (int k0 = 0; k0 < K; k0 += 32) {
        AFrag a, b;
        const unsigned short* ap = A + (long)mld * K + k0 + kb;
        if (k0 + 32 < K) __builtin_prefetch(ap + 32, 0, 1);   // global_prefetch_b8
#pragma unroll
        for (int i = 0; i < 4; ++i) {
            a.u32[i]     = *(const unsigned int*)(ap + 2 * i);        // K = kb+2i,+1
            a.u32[i + 4] = *(const unsigned int*)(ap + 16 + 2 * i);   // K = 16+kb+2i,+1
        }
#pragma unroll
        for (int t = 0; t < NT; ++t) {
            const unsigned short* wp = Wt + (long)(n0 + t * 16 + lcol) * K + k0 + kb;
#pragma unroll
            for (int i = 0; i < 4; ++i) {
                b.u32[i]     = *(const unsigned int*)(wp + 2 * i);
                b.u32[i + 4] = *(const unsigned int*)(wp + 16 + 2 * i);
            }
            acc[t] = __builtin_amdgcn_wmma_f32_16x16x32_bf16(
                         false, a.v, false, b.v, (short)0, acc[t], false, false);
        }
        if (DUAL) {
            const unsigned short* ap2 = A2 + (long)mld * K + k0 + kb;
#pragma unroll
            for (int i = 0; i < 4; ++i) {
                a.u32[i]     = *(const unsigned int*)(ap2 + 2 * i);
                a.u32[i + 4] = *(const unsigned int*)(ap2 + 16 + 2 * i);
            }
#pragma unroll
            for (int t = 0; t < NT; ++t) {
                const unsigned short* wp2 = W2t + (long)(n0 + t * 16 + lcol) * K + k0 + kb;
#pragma unroll
                for (int i = 0; i < 4; ++i) {
                    b.u32[i]     = *(const unsigned int*)(wp2 + 2 * i);
                    b.u32[i + 4] = *(const unsigned int*)(wp2 + 16 + 2 * i);
                }
                acc[t] = __builtin_amdgcn_wmma_f32_16x16x32_bf16(
                             false, a.v, false, b.v, (short)0, acc[t], false, false);
            }
        }
    }

#pragma unroll
    for (int t = 0; t < NT; ++t) {
        const int ncol = n0 + t * 16 + lcol;
        const float bv = bias[ncol];
#pragma unroll
        for (int r = 0; r < 8; ++r) {
            int row = m0 + r + ((lane >> 4) * 8);   // C/D: VGPR r -> M=r / M=8+r
            if (row < nrows) {
                long idx = (long)row * NOUT + ncol;
                float vo = acc[t][r] + bv;
                if (HASRES) vo += res[idx];
                if (RELU)   vo = fmaxf(vo, 0.0f);
                out[idx] = vo;
                if (STOREBF) outbf[idx] = f2bf(vo);
            }
        }
    }
}

// ---------------- attention kernels ----------------
// one wave per (edge, head): logits = (q[dst].k[src]) / sqrt(128); seg-max
__global__ __launch_bounds__(128)
void k_attn_logits(const float* __restrict__ q, const float* __restrict__ kk,
                   const int* __restrict__ src, const int* __restrict__ dst,
                   float* __restrict__ logits, float* __restrict__ mmax, int E)
{
    int e = blockIdx.x;
    if (e >= E) return;
    int head = threadIdx.y, lane = threadIdx.x;
    int s = src[e], d = dst[e];
    const float* qp = q  + (long)d * 512 + head * 128;
    const float* kp = kk + (long)s * 512 + head * 128;
    float sum = 0.0f;
#pragma unroll
    for (int i = 0; i < 4; ++i) {
        int c = lane + 32 * i;
        sum += qp[c] * kp[c];
    }
#pragma unroll
    for (int off = 16; off > 0; off >>= 1) sum += __shfl_xor(sum, off, 32);
    if (lane == 0) {
        float lg = sum * 0.08838834764831845f;   // 128^-0.5
        logits[(long)e * 4 + head] = lg;
        atomicMaxF(&mmax[(long)d * 4 + head], lg);
    }
}

// exp(logit - max) and segment-sum of denominators; alpha stored in-place
__global__ void k_attn_exp(const int* __restrict__ dst,
                           float* __restrict__ logits,
                           const float* __restrict__ mmax,
                           float* __restrict__ den, int E)
{
    long t = (long)blockIdx.x * blockDim.x + threadIdx.x;
    if (t >= (long)E * 4) return;
    int e = (int)(t >> 2), head = (int)(t & 3);
    int d = dst[e];
    float ex = __expf(logits[t] - mmax[(long)d * 4 + head]);
    logits[t] = ex;
    atomicAdd(&den[(long)d * 4 + head], ex);
}

// acc512[dst] += alpha * v[src]
__global__ __launch_bounds__(256)
void k_attn_apply(const float* __restrict__ vbuf,
                  const int* __restrict__ src, const int* __restrict__ dst,
                  const float* __restrict__ alpha, const float* __restrict__ den,
                  float* __restrict__ acc512, int E)
{
    int e = blockIdx.x;
    if (e >= E) return;
    int s = src[e], d = dst[e];
    for (int ch = threadIdx.x; ch < 512; ch += 256) {
        int head = ch >> 7;
        float w = alpha[(long)e * 4 + head] / (den[(long)d * 4 + head] + 1e-16f);
        atomicAdd(&acc512[(long)d * 512 + ch], w * vbuf[(long)s * 512 + ch]);
    }
}

// ---------------- pooling + fc ----------------
__global__ void k_pool(const float* __restrict__ h, const int* __restrict__ batch,
                       float* __restrict__ pooled, float* __restrict__ cnt, int n)
{
    int i = blockIdx.x;
    if (i >= n) return;
    int b = batch[i];
    atomicAdd(&pooled[(long)b * 128 + threadIdx.x], h[(long)i * 128 + threadIdx.x]);
    if (threadIdx.x == 0) atomicAdd(&cnt[b], 1.0f);
}

__global__ void k_fc(const float* __restrict__ pooled, const float* __restrict__ cnt,
                     const float* __restrict__ fcW, const float* __restrict__ fcb,
                     float* __restrict__ out)
{
    int c = threadIdx.x;   // 0..7
    int g = threadIdx.y;   // 0..63
    float inv = 1.0f / fmaxf(cnt[g], 1.0f);
    float s = fcb[c];
    for (int k = 0; k < 128; ++k)
        s += (pooled[(long)g * 128 + k] * inv) * fcW[(long)k * 8 + c];
    out[(long)g * 8 + c] = s;
}

// ---------------- host ----------------
extern "C" void kernel_launch(void* const* d_in, const int* in_sizes, int n_in,
                              void* d_out, int out_size, void* d_ws, size_t ws_size,
                              hipStream_t stream)
{
    const int N = in_sizes[2];          // batch has N entries
    const int E = in_sizes[1] / 2;
    const int G = 64, F = 128;

    const float* x       = (const float*)d_in[0];
    const int*   ei      = (const int*)  d_in[1];
    const int*   batch   = (const int*)  d_in[2];
    const float* c1_Wr   = (const float*)d_in[3];
    const float* c1_br   = (const float*)d_in[4];
    const float* c1_Wrt  = (const float*)d_in[5];
    const float* tqW = (const float*)d_in[6];  const float* tqb = (const float*)d_in[7];
    const float* tkW = (const float*)d_in[8];  const float* tkb = (const float*)d_in[9];
    const float* tvW = (const float*)d_in[10]; const float* tvb = (const float*)d_in[11];
    const float* tsW = (const float*)d_in[12]; const float* tsb = (const float*)d_in[13];
    const float* skW = (const float*)d_in[14];
    const float* l1W = (const float*)d_in[15]; const float* l1b = (const float*)d_in[16];
    const float* g1Wr = (const float*)d_in[17]; const float* g1br = (const float*)d_in[18];
    const float* g1Wrt= (const float*)d_in[19];
    const float* g2Wr = (const float*)d_in[20]; const float* g2br = (const float*)d_in[21];
    const float* g2Wrt= (const float*)d_in[22];
    const float* fcW  = (const float*)d_in[23]; const float* fcb = (const float*)d_in[24];

    const int* src = ei;
    const int* dst = ei + E;

    // ---- workspace carve-up ----
    size_t cur = 0;
    auto alloc = [&](size_t bytes) -> void* {
        cur = (cur + 255) & ~(size_t)255;
        void* p = (char*)d_ws + cur;
        cur += bytes;
        return p;
    };
    unsigned short* xbf   = (unsigned short*)alloc((size_t)N * F * 2);
    unsigned short* wbf   = (unsigned short*)alloc((size_t)491520 * 2);
    unsigned short* aggbf = (unsigned short*)alloc((size_t)N * 128 * 2);
    unsigned short* h1bf  = (unsigned short*)alloc((size_t)N * 128 * 2);
    unsigned short* h2bf  = (unsigned short*)alloc((size_t)N * 512 * 2);
    unsigned short* h3bf  = (unsigned short*)alloc((size_t)N * 128 * 2);
    unsigned short* h4bf  = (unsigned short*)alloc((size_t)N * 128 * 2);
    float* agg    = (float*)alloc((size_t)N * 128 * 4);
    float* h1     = (float*)alloc((size_t)N * 128 * 4);
    float* qb     = (float*)alloc((size_t)N * 512 * 4);
    float* kb     = (float*)alloc((size_t)N * 512 * 4);
    float* vb     = (float*)alloc((size_t)N * 512 * 4);
    float* acc512 = (float*)alloc((size_t)N * 512 * 4);
    float* lgts   = (float*)alloc((size_t)E * 4 * 4);
    float* mmax   = (float*)alloc((size_t)N * 4 * 4);
    float* den    = (float*)alloc((size_t)N * 4 * 4);
    float* h3     = (float*)alloc((size_t)N * 128 * 4);
    float* h4     = (float*)alloc((size_t)N * 128 * 4);
    float* h5     = (float*)alloc((size_t)N * 128 * 4);
    float* pooled = (float*)alloc((size_t)G * 128 * 4);
    float* cnt    = (float*)alloc((size_t)G * 4);

    // bf16 transposed weight slots ([NOUT, K] each)
    unsigned short* W_c1r  = wbf + 0;
    unsigned short* W_c1rt = wbf + 16384;
    unsigned short* W_tq   = wbf + 32768;
    unsigned short* W_tk   = wbf + 98304;
    unsigned short* W_tv   = wbf + 163840;
    unsigned short* W_ts   = wbf + 229376;
    unsigned short* W_sk   = wbf + 294912;
    unsigned short* W_l1   = wbf + 360448;
    unsigned short* W_g1r  = wbf + 425984;
    unsigned short* W_g1rt = wbf + 442368;
    unsigned short* W_g2r  = wbf + 458752;
    unsigned short* W_g2rt = wbf + 475136;

    auto cvt = [&](const float* s, unsigned short* dsts, long n) {
        k_f32_to_bf16<<<(unsigned)((n + 255) / 256), 256, 0, stream>>>(s, dsts, n);
    };
    auto cvtT = [&](const float* s, unsigned short* dsts, int K, int NOUT) {
        long n = (long)K * NOUT;
        k_f32_to_bf16_T<<<(unsigned)((n + 255) / 256), 256, 0, stream>>>(s, dsts, K, NOUT);
    };
    auto zero = [&](float* p, long n) {
        k_fill<<<(unsigned)((n + 255) / 256), 256, 0, stream>>>(p, 0.0f, n);
    };

    // --- convert inputs & transpose weights to bf16 ---
    cvt(x, xbf, (long)N * F);
    cvtT(c1_Wr, W_c1r, 128, 128);   cvtT(c1_Wrt, W_c1rt, 128, 128);
    cvtT(tqW, W_tq, 128, 512);      cvtT(tkW, W_tk, 128, 512);
    cvtT(tvW, W_tv, 128, 512);      cvtT(tsW, W_ts, 128, 512);
    cvtT(skW, W_sk, 128, 512);      cvtT(l1W, W_l1, 512, 128);
    cvtT(g1Wr, W_g1r, 128, 128);    cvtT(g1Wrt, W_g1rt, 128, 128);
    cvtT(g2Wr, W_g2r, 128, 128);    cvtT(g2Wrt, W_g2rt, 128, 128);

    const long NE128 = (long)E * 128;
    const unsigned scatGrid = (unsigned)((NE128 + 255) / 256);
    const dim3 gblk(32, 4);
    const unsigned mtiles = (unsigned)((N + 15) / 16);

    // --- GraphConv 1: agg = scatter(x); h1 = relu(agg@Wr + br + x@Wroot) ---
    zero(agg, (long)N * 128);
    k_edge_scatter128<<<scatGrid, 256, 0, stream>>>(x, src, dst, agg, E);
    cvt(agg, aggbf, (long)N * 128);
    k_wmma_gemm<128, 128, 2, true, true, false, true>
        <<<dim3(mtiles, 1), gblk, 0, stream>>>(
            aggbf, W_c1r, xbf, W_c1rt, c1_br, nullptr, h1, h1bf, N);

    // --- q, k, v projections (128 -> 512) ---
    k_wmma_gemm<128, 512, 4, false, false, false, false>
        <<<dim3(mtiles, 2), gblk, 0, stream>>>(
            h1bf, W_tq, nullptr, nullptr, tqb, nullptr, qb, nullptr, N);
    k_wmma_gemm<128, 512, 4, false, false, false, false>
        <<<dim3(mtiles, 2), gblk, 0, stream>>>(
            h1bf, W_tk, nullptr, nullptr, tkb, nullptr, kb, nullptr, N);
    k_wmma_gemm<128, 512, 4, false, false, false, false>
        <<<dim3(mtiles, 2), gblk, 0, stream>>>(
            h1bf, W_tv, nullptr, nullptr, tvb, nullptr, vb, nullptr, N);

    // --- acc512 = h1@tskip_W + tskip_b + h1@skip_W  (attention added on top) ---
    k_wmma_gemm<128, 512, 4, true, false, false, false>
        <<<dim3(mtiles, 2), gblk, 0, stream>>>(
            h1bf, W_ts, h1bf, W_sk, tsb, nullptr, acc512, nullptr, N);

    // --- edge softmax attention ---
    k_fill<<<(unsigned)(((long)N * 4 + 255) / 256), 256, 0, stream>>>(mmax, -INFINITY, (long)N * 4);
    zero(den, (long)N * 4);
    k_attn_logits<<<E, dim3(32, 4), 0, stream>>>(qb, kb, src, dst, lgts, mmax, E);
    k_attn_exp<<<(unsigned)(((long)E * 4 + 255) / 256), 256, 0, stream>>>(dst, lgts, mmax, den, E);
    k_attn_apply<<<E, 256, 0, stream>>>(vb, src, dst, lgts, den, acc512, E);

    // --- h2 = relu(t_out + skip_out) -> bf16 ; lin1: h3 = relu(h2@lin1_W + b) ---
    k_relu_to_bf16<<<(unsigned)(((long)N * 512 + 255) / 256), 256, 0, stream>>>(
        acc512, h2bf, (long)N * 512);
    k_wmma_gemm<512, 128, 2, false, true, false, true>
        <<<dim3(mtiles, 1), gblk, 0, stream>>>(
            h2bf, W_l1, nullptr, nullptr, l1b, nullptr, h3, h3bf, N);

    // --- GraphConv l1 (residual) ---
    zero(agg, (long)N * 128);
    k_edge_scatter128<<<scatGrid, 256, 0, stream>>>(h3, src, dst, agg, E);
    cvt(agg, aggbf, (long)N * 128);
    k_wmma_gemm<128, 128, 2, true, true, true, true>
        <<<dim3(mtiles, 1), gblk, 0, stream>>>(
            aggbf, W_g1r, h3bf, W_g1rt, g1br, h3, h4, h4bf, N);

    // --- GraphConv l2 (residual) ---
    zero(agg, (long)N * 128);
    k_edge_scatter128<<<scatGrid, 256, 0, stream>>>(h4, src, dst, agg, E);
    cvt(agg, aggbf, (long)N * 128);
    k_wmma_gemm<128, 128, 2, true, true, true, false>
        <<<dim3(mtiles, 1), gblk, 0, stream>>>(
            aggbf, W_g2r, h4bf, W_g2rt, g2br, h4, h5, nullptr, N);

    // --- global mean pool + fc ---
    zero(pooled, (long)G * 128);
    zero(cnt, G);
    k_pool<<<N, 128, 0, stream>>>(h5, batch, pooled, cnt, N);
    k_fc<<<1, dim3(8, 64), 0, stream>>>(pooled, cnt, fcW, fcb, (float*)d_out);
}
